// TensorNet_58145267253390
// MI455X (gfx1250) — matrix operand order repaired and checked by reading, not compiled, not run on hardware
//
#include <hip/hip_runtime.h>
#include <math.h>

// ---------------------------------------------------------------------------
// TensorNet forward for MI455X (gfx1250, wave32, WMMA).
// Dense linears: v_wmma_f32_16x16x32_f16, A read as f32 and converted
// in-register (v_cvt co-executes with the XDL pipe), W pre-converted to f16.
// Each wave computes a 16x64 output strip (1 M-tile x 4 N-tiles) so the A
// fragment is reused across 4 WMMAs per K-step.
// Node tensor layout internally: [N, 3, 3, C]  (so _chan == contiguous GEMM).
// ---------------------------------------------------------------------------

static const int N_ = 10000;
static const int E_ = 160000;
static const int C_ = 128;

typedef _Float16 half_t;
typedef __attribute__((ext_vector_type(16))) _Float16 v16h;
typedef __attribute__((ext_vector_type(8)))  _Float16 v8h;
typedef __attribute__((ext_vector_type(8)))  float    v8f;
typedef __attribute__((ext_vector_type(4)))  float    v4f;

__device__ __forceinline__ float cutoff_f(float d) {
  float c = 0.5f * (__cosf(d * 0.6283185307179586f) + 1.f); // pi/5
  return d < 5.0f ? c : 0.f;
}

// ---------------------------------------------------------------------------
// WMMA GEMM: out[M,Nout] = act(A[M,Kp] @ W[Nout,Kp]^T + bias)
// A row-major f32 (converted in-register), W row-major f16.
// Kp % 32 == 0, M % 16 == 0, Nout % 64 == 0.
// One 16x64 strip per wave32 (4 accumulators); 8 waves per 256-thread block.
// ---------------------------------------------------------------------------
__global__ void gemm_wmma_af32(const float*  __restrict__ A,
                               const half_t* __restrict__ W,
                               const float*  __restrict__ bias,
                               float* __restrict__ out,
                               int M, int Kp, int Nout, int act)
{
  int wave = (int)((blockIdx.x * blockDim.x + threadIdx.x) >> 5);
  int lane = threadIdx.x & 31;
  int ng = Nout >> 6;                       // groups of 4 N-tiles
  int groups = (M >> 4) * ng;
  if (wave >= groups) return;               // wave-uniform: EXEC stays all-1s
  int mT = wave / ng;
  int nG = wave - mT * ng;
  int g   = lane >> 4;                      // half-wave group
  int l16 = lane & 15;
  const float*  arow  = A + (size_t)(mT * 16 + l16) * Kp;
  const half_t* wrow  = W + (size_t)(nG * 64 + l16) * Kp;
  const size_t  wstep = (size_t)16 * Kp;
  v8f acc[4] = {};
  for (int kk = 0; kk < Kp; kk += 32) {
    if (kk + 32 < Kp)
      __builtin_prefetch(arow + kk + 32, 0, 1);     // global_prefetch_b8
    // A 16x32 f16 fragment: halves 0..7 -> K = kk + g*8 + h,
    //                       halves 8..15 -> K = kk + 16 + g*8 + (h-8)
    v4f f0 = *(const v4f*)(arow + kk + g * 8);
    v4f f1 = *(const v4f*)(arow + kk + g * 8 + 4);
    v4f f2 = *(const v4f*)(arow + kk + 16 + g * 8);
    v4f f3 = *(const v4f*)(arow + kk + 16 + g * 8 + 4);
    v16h av;
#pragma unroll
    for (int i = 0; i < 4; ++i) {
      av[i]      = (half_t)f0[i];
      av[i + 4]  = (half_t)f1[i];
      av[i + 8]  = (half_t)f2[i];
      av[i + 12] = (half_t)f3[i];
    }
#pragma unroll
    for (int s = 0; s < 4; ++s) {
      // B 32x16 f16 fragment (W row-major [Nout,K]): same K order as A
      const half_t* wr = wrow + (size_t)s * wstep;
      v8h b0 = *(const v8h*)(wr + kk + g * 8);
      v8h b1 = *(const v8h*)(wr + kk + 16 + g * 8);
      v16h bv;
#pragma unroll
      for (int i = 0; i < 8; ++i) { bv[i] = b0[i]; bv[i + 8] = b1[i]; }
      acc[s] = __builtin_amdgcn_wmma_f32_16x16x32_f16(
          /*neg_a=*/false, av, /*neg_b=*/false, bv,
          /*c_mod=*/(short)0, acc[s], /*reuse_a=*/false, /*reuse_b=*/false);
    }
  }
#pragma unroll
  for (int s = 0; s < 4; ++s) {
    int ncol = nG * 64 + s * 16 + l16;
    float bb = bias ? bias[ncol] : 0.f;
#pragma unroll
    for (int r = 0; r < 8; ++r) {           // C/D layout: m = r + 8*g, n = lane&15
      float v = acc[s][r] + bb;
      if (act) v = v / (1.f + __expf(-v));  // SiLU
      int m = mT * 16 + g * 8 + r;
      out[(size_t)m * Nout + ncol] = v;
    }
  }
}

// --------------------------- support kernels -------------------------------

// f32 -> f16 with zero-padding of the inner dim (Ks -> Kd) — weights only
__global__ void k_cvt_f16(const float* __restrict__ src, half_t* __restrict__ dst,
                          int M, int Ks, int Kd)
{
  int i = blockIdx.x * 256 + threadIdx.x;
  if (i >= M * Kd) return;
  int k = i % Kd;
  int m = i / Kd;
  dst[i] = (k < Ks) ? (half_t)src[(size_t)m * Ks + k] : (half_t)0.f;
}

// h = concat(emb[z], t) zero-padded to 160, f32
__global__ void k_build_h(const int* __restrict__ z, const float* __restrict__ t,
                          const float* __restrict__ emb, float* __restrict__ h, int N)
{
  int i = blockIdx.x * 256 + threadIdx.x;
  if (i >= N * 160) return;
  int k = i % 160, n = i / 160;
  float v = 0.f;
  if (k < 128)       v = emb[(size_t)z[n] * 128 + k];
  else if (k == 128) v = t[n];
  h[i] = v;
}

// Zpair f32 [E,256] = concat(Z[src], Z[dst])
__global__ void k_zpair(const float* __restrict__ Z, const int* __restrict__ ei,
                        float* __restrict__ zp, int E)
{
  int i = blockIdx.x * 256 + threadIdx.x;
  if (i >= E * 256) return;
  int k = i & 255, e = i >> 8;
  int n = (k < 128) ? ei[e] : ei[E + e];
  zp[i] = Z[(size_t)n * 128 + (k & 127)];
}

// Embedding scatter, compressed: I scalar, A 3-vec (skew is linear),
// S 6 sym components. 10 atomics per (edge, channel).
__global__ void k_edge_scatter(const float* __restrict__ G1, const float* __restrict__ G2,
                               const float* __restrict__ G3, const float* __restrict__ Zij,
                               const float* __restrict__ ew, const float* __restrict__ evn,
                               const int* __restrict__ ei,
                               float* accI, float* accA, float* accS, int E)
{
  int e = blockIdx.x, c = threadIdx.x;
  float cut = cutoff_f(ew[e]);
  float zb  = cut * Zij[(size_t)e * 128 + c];
  float dI  = G1[(size_t)e * 128 + c] * zb;
  float dA  = G2[(size_t)e * 128 + c] * zb;
  float dS  = G3[(size_t)e * 128 + c] * zb;
  int   n   = ei[e];
  float vx = evn[e * 3 + 0], vy = evn[e * 3 + 1], vz = evn[e * 3 + 2];
  float third = (vx * vx + vy * vy + vz * vz) * (1.f / 3.f);
  int nc = n * 128 + c;
  atomicAdd(&accI[nc], dI);
  atomicAdd(&accA[nc * 3 + 0], dA * vx);
  atomicAdd(&accA[nc * 3 + 1], dA * vy);
  atomicAdd(&accA[nc * 3 + 2], dA * vz);
  atomicAdd(&accS[nc * 6 + 0], dS * (vx * vx - third));
  atomicAdd(&accS[nc * 6 + 1], dS * (vx * vy));
  atomicAdd(&accS[nc * 6 + 2], dS * (vx * vz));
  atomicAdd(&accS[nc * 6 + 3], dS * (vy * vy - third));
  atomicAdd(&accS[nc * 6 + 4], dS * (vy * vz));
  atomicAdd(&accS[nc * 6 + 5], dS * (vz * vz - third));
}

// Expand compressed accumulators into I/A/S [N,3,3,C], plus tensor_norm(I+A+S)
__global__ void k_expand(const float* __restrict__ accI, const float* __restrict__ accA,
                         const float* __restrict__ accS,
                         float* I, float* A, float* S, float* normRaw)
{
  int n = blockIdx.x, c = threadIdx.x;
  int nc = n * 128 + c;
  float iv = accI[nc];
  float a0 = accA[nc * 3 + 0], a1 = accA[nc * 3 + 1], a2 = accA[nc * 3 + 2];
  float s0 = accS[nc * 6 + 0], s1 = accS[nc * 6 + 1], s2 = accS[nc * 6 + 2];
  float s3 = accS[nc * 6 + 3], s4 = accS[nc * 6 + 4], s5 = accS[nc * 6 + 5];
  float Im[9] = { iv, 0, 0,   0, iv, 0,   0, 0, iv };
  float Am[9] = { 0, -a2, a1,  a2, 0, -a0,  -a1, a0, 0 };
  float Sm[9] = { s0, s1, s2,  s1, s3, s4,  s2, s4, s5 };
  float ss = 0.f;
  size_t base = (size_t)n * 1152 + c;
#pragma unroll
  for (int q = 0; q < 9; ++q) {
    float tv = Im[q] + Am[q] + Sm[q];
    ss += tv * tv;
    I[base + q * 128] = Im[q];
    A[base + q * 128] = Am[q];
    S[base + q * 128] = Sm[q];
  }
  normRaw[nc] = fmaxf(ss, 0.01f);
}

// LayerNorm over C=128, one node per 128-thread block, LDS tree reduction
__global__ void k_layernorm(const float* __restrict__ x, const float* __restrict__ g,
                            const float* __restrict__ b, float* __restrict__ y)
{
  __shared__ float sm[128];
  int n = blockIdx.x, c = threadIdx.x;
  float v = x[n * 128 + c];
  sm[c] = v; __syncthreads();
  for (int s = 64; s > 0; s >>= 1) { if (c < s) sm[c] += sm[c + s]; __syncthreads(); }
  float mean = sm[0] * (1.f / 128.f);
  __syncthreads();
  float d = v - mean;
  sm[c] = d * d; __syncthreads();
  for (int s = 64; s > 0; s >>= 1) { if (c < s) sm[c] += sm[c + s]; __syncthreads(); }
  float var = sm[0] * (1.f / 128.f);
  y[n * 128 + c] = d * rsqrtf(var + 1e-5f) * g[c] + b[c];
}

// X = cI*norm3[...,0] + cA*norm3[...,1] + cS*norm3[...,2]
__global__ void k_combine(const float* __restrict__ cI, const float* __restrict__ cA,
                          const float* __restrict__ cS, const float* __restrict__ norm3,
                          float* __restrict__ X, int total)
{
  int i = blockIdx.x * 256 + threadIdx.x;
  if (i >= total) return;
  int c = i & 127;
  int n = (i >> 7) / 9;
  const float* nr = norm3 + (size_t)n * 384 + c * 3;
  X[i] = cI[i] * nr[0] + cA[i] * nr[1] + cS[i] * nr[2];
}

// Xn = Xp / (tensor_norm(Xp)+1), then decompose into I/A/S
__global__ void k_norm_decomp(const float* __restrict__ Xp, float* Xn,
                              float* I, float* A, float* S)
{
  int n = blockIdx.x, c = threadIdx.x;
  size_t base = (size_t)n * 1152 + c;
  float x[9]; float ss = 0.f;
#pragma unroll
  for (int q = 0; q < 9; ++q) { x[q] = Xp[base + q * 128]; ss += x[q] * x[q]; }
  float inv = 1.f / (fmaxf(ss, 0.01f) + 1.f);
#pragma unroll
  for (int q = 0; q < 9; ++q) { x[q] *= inv; Xn[base + q * 128] = x[q]; }
  float iv = (x[0] + x[4] + x[8]) * (1.f / 3.f);
#pragma unroll
  for (int i3 = 0; i3 < 3; ++i3)
#pragma unroll
    for (int j3 = 0; j3 < 3; ++j3) {
      int q = i3 * 3 + j3, qt = j3 * 3 + i3;
      float sym = 0.5f * (x[q] + x[qt]);
      float dia = (i3 == j3) ? iv : 0.f;
      I[base + q * 128] = dia;
      A[base + q * 128] = 0.5f * (x[q] - x[qt]);
      S[base + q * 128] = sym - dia;
    }
}

__global__ void k_add3(const float* __restrict__ a, const float* __restrict__ b,
                       const float* __restrict__ c, float* __restrict__ o,
                       int total, int accum)
{
  int i = blockIdx.x * 256 + threadIdx.x;
  if (i >= total) return;
  float v = a[i] + b[i] + c[i];
  o[i] = accum ? o[i] + v : v;
}

// Gather I'/A'/S' at dst, scale by h(=h3*cutoff), scatter-add into msg at src.
// Node tensors (3 x 46MB + msg) are L2-resident on MI455X (192MB L2).
__global__ void k_msgpass(const float* __restrict__ h3, const float* __restrict__ ew,
                          const int* __restrict__ ei, int E,
                          const float* __restrict__ Ip, const float* __restrict__ Ap,
                          const float* __restrict__ Sp, float* __restrict__ msg)
{
  int e = blockIdx.x, d = threadIdx.x;
  int src = ei[e], dst = ei[E + e];
  float cut = cutoff_f(ew[e]);
  const float* hp = h3 + (size_t)e * 384 + d * 3;
  float h0 = hp[0] * cut, h1 = hp[1] * cut, h2 = hp[2] * cut;
  size_t bs = (size_t)dst * 1152 + d;
  size_t bd = (size_t)src * 1152 + d;
#pragma unroll
  for (int q = 0; q < 9; ++q) {
    float val = h0 * Ip[bs + q * 128] + h1 * Ap[bs + q * 128] + h2 * Sp[bs + q * 128];
    atomicAdd(&msg[bd + q * 128], val);
  }
}

// T = msg@Y + Y@msg, decompose, scale by 1/(tensor_norm(T)+1)
__global__ void k_o3(const float* __restrict__ msg, const float* __restrict__ Y,
                     float* I, float* A, float* S)
{
  int n = blockIdx.x, c = threadIdx.x;
  size_t base = (size_t)n * 1152 + c;
  float m[9], y[9];
#pragma unroll
  for (int q = 0; q < 9; ++q) { m[q] = msg[base + q * 128]; y[q] = Y[base + q * 128]; }
  float tm[9]; float ss = 0.f;
#pragma unroll
  for (int i3 = 0; i3 < 3; ++i3)
#pragma unroll
    for (int j3 = 0; j3 < 3; ++j3) {
      float a2 = 0.f, b2 = 0.f;
#pragma unroll
      for (int k3 = 0; k3 < 3; ++k3) {
        a2 += m[i3 * 3 + k3] * y[k3 * 3 + j3];
        b2 += y[i3 * 3 + k3] * m[k3 * 3 + j3];
      }
      tm[i3 * 3 + j3] = a2 + b2;
    }
#pragma unroll
  for (int q = 0; q < 9; ++q) ss += tm[q] * tm[q];
  float inv = 1.f / (fmaxf(ss, 0.01f) + 1.f);
  float iv  = (tm[0] + tm[4] + tm[8]) * (1.f / 3.f);
#pragma unroll
  for (int i3 = 0; i3 < 3; ++i3)
#pragma unroll
    for (int j3 = 0; j3 < 3; ++j3) {
      int q = i3 * 3 + j3, qt = j3 * 3 + i3;
      float sym = 0.5f * (tm[q] + tm[qt]);
      float dia = (i3 == j3) ? iv : 0.f;
      I[base + q * 128] = dia * inv;
      A[base + q * 128] = 0.5f * (tm[q] - tm[qt]) * inv;
      S[base + q * 128] = (sym - dia) * inv;
    }
}

// [N,3,3,C] -> [N,C,3,3]
__global__ void k_transpose_out(const float* __restrict__ X, float* __restrict__ out, int total)
{
  int i = blockIdx.x * 256 + threadIdx.x;
  if (i >= total) return;
  int q = i % 9;
  int c = (i / 9) % 128;
  int n = i / (9 * 128);
  out[i] = X[((size_t)n * 9 + q) * 128 + c];
}

// ---------------------------------------------------------------------------

extern "C" void kernel_launch(void* const* d_in, const int* in_sizes, int n_in,
                              void* d_out, int out_size, void* d_ws, size_t ws_size,
                              hipStream_t stream)
{
  (void)in_sizes; (void)n_in; (void)out_size; (void)ws_size;

  const int*   z   = (const int*)  d_in[0];
  const float* t   = (const float*)d_in[1];
  const int*   ei  = (const int*)  d_in[2];      // [2,E] row-major
  const float* ew  = (const float*)d_in[3];
  const float* evn = (const float*)d_in[4];
  const float* ea  = (const float*)d_in[5];      // [E,64] — direct GEMM A operand
  const float* emb    = (const float*)d_in[6];
  const float* mix_w1 = (const float*)d_in[7];  const float* mix_b1 = (const float*)d_in[8];
  const float* mix_w2 = (const float*)d_in[9];  const float* mix_b2 = (const float*)d_in[10];
  const float* emb2_w = (const float*)d_in[11]; const float* emb2_b = (const float*)d_in[12];
  const float* dp_w[3] = {(const float*)d_in[13], (const float*)d_in[15], (const float*)d_in[17]};
  const float* dp_b[3] = {(const float*)d_in[14], (const float*)d_in[16], (const float*)d_in[18]};
  const float* ln_g   = (const float*)d_in[19]; const float* ln_b   = (const float*)d_in[20];
  const float* esc1_w = (const float*)d_in[21]; const float* esc1_b = (const float*)d_in[22];
  const float* esc2_w = (const float*)d_in[23]; const float* esc2_b = (const float*)d_in[24];
  const float* elt    = (const float*)d_in[25];   // [3,128,128]
  const float *sc1_w[2], *sc1_b[2], *sc2_w[2], *sc2_b[2], *sc3_w[2], *sc3_b[2],
              *proj_w[2], *proj_b[2], *ltp[2];
  for (int L = 0; L < 2; ++L) {
    int b = 26 + 9 * L;
    sc1_w[L]  = (const float*)d_in[b + 0]; sc1_b[L]  = (const float*)d_in[b + 1];
    sc2_w[L]  = (const float*)d_in[b + 2]; sc2_b[L]  = (const float*)d_in[b + 3];
    sc3_w[L]  = (const float*)d_in[b + 4]; sc3_b[L]  = (const float*)d_in[b + 5];
    proj_w[L] = (const float*)d_in[b + 6]; proj_b[L] = (const float*)d_in[b + 7];
    ltp[L]    = (const float*)d_in[b + 8];          // [6,128,128]
  }

  // ---- workspace bump allocator (256B aligned) ----
  char* wsb = (char*)d_ws;
  size_t off = 0;
  auto alloc = [&](size_t bytes) -> void* {
    void* p = wsb + off;
    off += (bytes + 255) & ~(size_t)255;
    return p;
  };
  const size_t NC9 = (size_t)N_ * 9 * C_;           // 11.52M elems / 46.1MB f32
  float* bufX   = (float*)alloc(NC9 * 4);
  float* bufP   = (float*)alloc(NC9 * 4);
  float* bufI   = (float*)alloc(NC9 * 4);
  float* bufA   = (float*)alloc(NC9 * 4);
  float* bufS   = (float*)alloc(NC9 * 4);
  float* bufIp  = (float*)alloc(NC9 * 4);
  float* bufAp  = (float*)alloc(NC9 * 4);
  float* bufSp  = (float*)alloc(NC9 * 4);
  float* bufY   = (float*)alloc(NC9 * 4);
  float* bufMsg = (float*)alloc(NC9 * 4);
  float* h1   = (float*)alloc((size_t)E_ * 128 * 4);
  float* h2   = (float*)alloc((size_t)E_ * 256 * 4);
  float* h3   = (float*)alloc((size_t)E_ * 384 * 4);
  float* Zij  = (float*)alloc((size_t)E_ * 128 * 4);
  float* Zbuf    = (float*)alloc((size_t)N_ * 128 * 4);
  float* Z1      = (float*)alloc((size_t)N_ * 128 * 4);
  float* normRaw = (float*)alloc((size_t)N_ * 128 * 4);
  float* normLN  = (float*)alloc((size_t)N_ * 128 * 4);
  float* norm3   = (float*)alloc((size_t)N_ * 384 * 4);
  float* esc1o   = (float*)alloc((size_t)N_ * 256 * 4);
  float* accI = (float*)alloc((size_t)N_ * 128 * 4);
  float* accA = (float*)alloc((size_t)N_ * 384 * 4);
  float* accS = (float*)alloc((size_t)N_ * 768 * 4);
  // aliases: used only before h1/h2 get their real contents
  float* hpad   = h1;   // [N,160] f32 padded concat(emb,t)   (N*160 < E*128)
  float* zpairF = h2;   // [E,256] f32 Zpair                  (exactly E*256)
  // f16 weight copies (K padded to %32)
  half_t* w_mix1 = (half_t*)alloc(128 * 160 * 2);
  half_t* w_mix2 = (half_t*)alloc(128 * 128 * 2);
  half_t* w_emb2 = (half_t*)alloc(128 * 256 * 2);
  half_t* w_dp[3];  for (int i = 0; i < 3; ++i) w_dp[i]  = (half_t*)alloc(128 * 64 * 2);
  half_t* w_esc1 = (half_t*)alloc(256 * 128 * 2);
  half_t* w_esc2 = (half_t*)alloc(384 * 256 * 2);
  half_t* w_elt[3]; for (int i = 0; i < 3; ++i) w_elt[i] = (half_t*)alloc(128 * 128 * 2);
  half_t *w_sc1[2], *w_sc2[2], *w_sc3[2], *w_proj[2], *w_lt[2][6];
  for (int L = 0; L < 2; ++L) {
    w_sc1[L]  = (half_t*)alloc(128 * 64 * 2);
    w_sc2[L]  = (half_t*)alloc(256 * 128 * 2);
    w_sc3[L]  = (half_t*)alloc(384 * 256 * 2);
    w_proj[L] = (half_t*)alloc(128 * 128 * 2);
    for (int i = 0; i < 6; ++i) w_lt[L][i] = (half_t*)alloc(128 * 128 * 2);
  }

  auto cvt = [&](const float* src, half_t* dst, int rows, int Ks, int Kd) {
    int total = rows * Kd;
    k_cvt_f16<<<(total + 255) / 256, 256, 0, stream>>>(src, dst, rows, Ks, Kd);
  };
  auto gemm = [&](const float* A, const half_t* W, const float* b, float* o,
                  int M, int Kp, int Nout, int act) {
    int groups = (M / 16) * (Nout / 64);
    gemm_wmma_af32<<<(groups + 7) / 8, 256, 0, stream>>>(A, W, b, o, M, Kp, Nout, act);
  };

  // ---- weight conversion (every launch; deterministic) ----
  cvt(mix_w1, w_mix1, 128, 129, 160);
  cvt(mix_w2, w_mix2, 128, 128, 128);
  cvt(emb2_w, w_emb2, 128, 256, 256);
  for (int i = 0; i < 3; ++i) cvt(dp_w[i], w_dp[i], 128, 64, 64);
  cvt(esc1_w, w_esc1, 256, 128, 128);
  cvt(esc2_w, w_esc2, 384, 256, 256);
  for (int i = 0; i < 3; ++i) cvt(elt + (size_t)i * 128 * 128, w_elt[i], 128, 128, 128);
  for (int L = 0; L < 2; ++L) {
    cvt(sc1_w[L],  w_sc1[L],  128, 64, 64);
    cvt(sc2_w[L],  w_sc2[L],  256, 128, 128);
    cvt(sc3_w[L],  w_sc3[L],  384, 256, 256);
    cvt(proj_w[L], w_proj[L], 128, 128, 128);
    for (int i = 0; i < 6; ++i) cvt(ltp[L] + (size_t)i * 128 * 128, w_lt[L][i], 128, 128, 128);
  }

  // ================= Embedding =================
  { int tot = N_ * 160; k_build_h<<<(tot + 255) / 256, 256, 0, stream>>>(z, t, emb, hpad, N_); }
  gemm(hpad, w_mix1, mix_b1, Z1, N_, 160, 128, 1);             // silu
  gemm(Z1,   w_mix2, mix_b2, Zbuf, N_, 128, 128, 0);
  { int tot = E_ * 256; k_zpair<<<(tot + 255) / 256, 256, 0, stream>>>(Zbuf, ei, zpairF, E_); }
  gemm(zpairF, w_emb2, emb2_b, Zij, E_, 256, 128, 0);
  gemm(ea, w_dp[0], dp_b[0], h1, E_, 64, 128, 0);              // overwrites hpad alias (done)
  gemm(ea, w_dp[1], dp_b[1], h2, E_, 64, 128, 0);              // overwrites zpair alias (done)
  gemm(ea, w_dp[2], dp_b[2], h3, E_, 64, 128, 0);
  hipMemsetAsync(accI, 0, (size_t)N_ * 128 * 4, stream);
  hipMemsetAsync(accA, 0, (size_t)N_ * 384 * 4, stream);
  hipMemsetAsync(accS, 0, (size_t)N_ * 768 * 4, stream);
  k_edge_scatter<<<E_, 128, 0, stream>>>(h1, h2, h3, Zij, ew, evn, ei, accI, accA, accS, E_);
  k_expand<<<N_, 128, 0, stream>>>(accI, accA, accS, bufI, bufA, bufS, normRaw);
  k_layernorm<<<N_, 128, 0, stream>>>(normRaw, ln_g, ln_b, normLN);
  gemm(normLN, w_esc1, esc1_b, esc1o, N_, 128, 256, 1);        // silu
  gemm(esc1o,  w_esc2, esc2_b, norm3, N_, 256, 384, 1);        // silu
  gemm(bufI, w_elt[0], nullptr, bufIp, 9 * N_, 128, 128, 0);
  gemm(bufA, w_elt[1], nullptr, bufAp, 9 * N_, 128, 128, 0);
  gemm(bufS, w_elt[2], nullptr, bufSp, 9 * N_, 128, 128, 0);
  { int tot = (int)NC9;
    k_combine<<<(tot + 255) / 256, 256, 0, stream>>>(bufIp, bufAp, bufSp, norm3, bufX, tot); }

  // ================= Interaction layers =================
  for (int L = 0; L < 2; ++L) {
    gemm(bufX, w_proj[L], proj_b[L], bufP, 9 * N_, 128, 128, 0);   // chan + bias
    k_norm_decomp<<<N_, 128, 0, stream>>>(bufP, bufX, bufI, bufA, bufS);
    // edge MLP
    gemm(ea, w_sc1[L], sc1_b[L], h1, E_, 64, 128, 1);
    gemm(h1, w_sc2[L], sc2_b[L], h2, E_, 128, 256, 1);
    gemm(h2, w_sc3[L], sc3_b[L], h3, E_, 256, 384, 1);
    // lt0..2 chan
    gemm(bufI, w_lt[L][0], nullptr, bufIp, 9 * N_, 128, 128, 0);
    gemm(bufA, w_lt[L][1], nullptr, bufAp, 9 * N_, 128, 128, 0);
    gemm(bufS, w_lt[L][2], nullptr, bufSp, 9 * N_, 128, 128, 0);
    { int tot = (int)NC9;
      k_add3<<<(tot + 255) / 256, 256, 0, stream>>>(bufIp, bufAp, bufSp, bufY, tot, 0); }
    // message passing
    hipMemsetAsync(bufMsg, 0, NC9 * 4, stream);
    k_msgpass<<<E_, 128, 0, stream>>>(h3, ew, ei, E_, bufIp, bufAp, bufSp, bufMsg);
    // O(3) product, decompose, normalize
    k_o3<<<N_, 128, 0, stream>>>(bufMsg, bufY, bufI, bufA, bufS);
    // lt3..5 chan
    gemm(bufI, w_lt[L][3], nullptr, bufIp, 9 * N_, 128, 128, 0);
    gemm(bufA, w_lt[L][4], nullptr, bufAp, 9 * N_, 128, 128, 0);
    gemm(bufS, w_lt[L][5], nullptr, bufSp, 9 * N_, 128, 128, 0);
    // residual: X = Xn + (I3+A3+S3)
    { int tot = (int)NC9;
      k_add3<<<(tot + 255) / 256, 256, 0, stream>>>(bufIp, bufAp, bufSp, bufX, tot, 1); }
  }

  // ================= Output transpose [N,3,3,C] -> [N,C,3,3] =================
  { int tot = (int)NC9;
    k_transpose_out<<<(tot + 255) / 256, 256, 0, stream>>>(bufX, (float*)d_out, tot); }
}
